// SymileLoss_21638045237260
// MI455X (gfx1250) — matrix operand order repaired and checked by reading, not compiled, not run on hardware
//
#include <hip/hip_runtime.h>
#include <hip/hip_bf16.h>
#include <math.h>

typedef __attribute__((ext_vector_type(16))) _Float16 v16h;
typedef __attribute__((ext_vector_type(8)))  float    v8f;
typedef __attribute__((ext_vector_type(4)))  _Float16 h4;

#define BSZ   256
#define DDIM  512
#define KT    128      // k-chunk staged in LDS
#define ROWS  128      // output rows per block (8 tile-rows)
#define NPR   (BSZ / ROWS)   // 2 row-chunks
#define LSTR  136      // halfs per LDS row (128 + 8 pad; 272B keeps 16B alignment)

#define XTILE_HALFS (ROWS * LSTR)                  // 17408 halfs
#define TILES_BYTES ((ROWS + BSZ) * LSTR * 2)      // 104448 B
#define M_BYTES     (ROWS * BSZ * 4)               // 131072 B
#define RP_OFF      M_BYTES                        // row pairs after M overlay
#define SMEM_BYTES  (M_BYTES + ROWS * 8)           // 132096 B

union FragU { v16h v; uint4 u[2]; };

__device__ inline void lse_merge(float2& a, float m, float s) {
  if (m > a.x) { a.y = a.y * __expf(a.x - m) + s; a.x = m; }
  else         { a.y += s * __expf(m - a.x); }
}

// One block per (row-chunk pr, shift s): M = T[pr*128..+128, :, s] via WMMA,
// then per-row / per-col / whole-block logsumexp partials.
__global__ __launch_bounds__(256)
void symile_tile_kernel(const float* __restrict__ x, const float* __restrict__ y,
                        const float* __restrict__ z,
                        float2* __restrict__ Apart,   // [s][p]      (256*256)
                        float2* __restrict__ Bpart,   // [s][pr][q]  (256*2*256)
                        float2* __restrict__ Cpart) { // [s][pr]     (256*2)
  extern __shared__ unsigned char smem[];
  _Float16* xs  = (_Float16*)smem;                   // ROWS x LSTR
  _Float16* wsm = xs + XTILE_HALFS;                  // BSZ  x LSTR
  float*    Ms  = (float*)smem;                      // overlay: ROWS x BSZ f32
  float2*   rp  = (float2*)(smem + RP_OFF);          // ROWS row pairs

  const int tid  = threadIdx.x;
  const int pr   = blockIdx.x;       // 0..1
  const int s    = blockIdx.y;       // 0..255
  const int rowbase = pr * ROWS;

  const int w    = tid >> 5;         // wave 0..7
  const int lane = tid & 31;
  const int lo   = lane & 15;
  const int hi   = lane >> 4;
  const int tr0  = (w >> 1) * 2;     // this wave's two tile-rows: tr0, tr0+1
  const int qbase = (w & 1) * 128;   // this wave's 8 tile-cols

  v8f acc0[8], acc1[8];
#pragma unroll
  for (int t = 0; t < 8; ++t) { acc0[t] = (v8f)0.0f; acc1[t] = (v8f)0.0f; }

  const float* zrow = z + (size_t)s * DDIM;

  for (int kc = 0; kc < DDIM / KT; ++kc) {
    __syncthreads();   // previous chunk's compute done before overwriting tiles
    // ---- stage X tile: 128 x 128 halfs (4096 h4 slots, 16/thread, batched 4) ----
#pragma unroll
    for (int g = 0; g < 4; ++g) {
      float4 v[4];
#pragma unroll
      for (int j = 0; j < 4; ++j) {
        int idx = tid + (g * 4 + j) * 256;
        int r = idx >> 5, k4 = idx & 31;
        v[j] = *(const float4*)(x + (size_t)(rowbase + r) * DDIM + kc * KT + k4 * 4);
      }
#pragma unroll
      for (int j = 0; j < 4; ++j) {
        int idx = tid + (g * 4 + j) * 256;
        int r = idx >> 5, k4 = idx & 31;
        h4 h = { (_Float16)v[j].x, (_Float16)v[j].y, (_Float16)v[j].z, (_Float16)v[j].w };
        *(h4*)(xs + r * LSTR + k4 * 4) = h;
      }
    }
    // ---- stage W tile: W[q,k] = y[q,k]*z[s,k], 256 x 128 halfs (32 slots/thread) ----
#pragma unroll
    for (int g = 0; g < 8; ++g) {
      float4 vy[4], vz[4];
#pragma unroll
      for (int j = 0; j < 4; ++j) {
        int idx = tid + (g * 4 + j) * 256;
        int q = idx >> 5, k4 = idx & 31;
        vy[j] = *(const float4*)(y + (size_t)q * DDIM + kc * KT + k4 * 4);
        vz[j] = *(const float4*)(zrow + kc * KT + k4 * 4);
      }
#pragma unroll
      for (int j = 0; j < 4; ++j) {
        int idx = tid + (g * 4 + j) * 256;
        int q = idx >> 5, k4 = idx & 31;
        h4 h = { (_Float16)(vy[j].x * vz[j].x), (_Float16)(vy[j].y * vz[j].y),
                 (_Float16)(vy[j].z * vz[j].z), (_Float16)(vy[j].w * vz[j].w) };
        *(h4*)(wsm + q * LSTR + k4 * 4) = h;
      }
    }
    __syncthreads();
    // ---- WMMA: 4 k-steps of 32 over this chunk ----
#pragma unroll
    for (int ks = 0; ks < 4; ++ks) {
      // Two A fragments (16x32 f16): lane holds row M=lo;
      // halfs[0..7]=K base+0..7, [8..15]=K base+16..23
      FragU af0, af1;
      {
        const _Float16* a0 = xs + (tr0 * 16 + lo) * LSTR + ks * 32 + hi * 8;
        const _Float16* a1 = a0 + 16 * LSTR;
        af0.u[0] = *(const uint4*)(a0);  af0.u[1] = *(const uint4*)(a0 + 16);
        af1.u[0] = *(const uint4*)(a1);  af1.u[1] = *(const uint4*)(a1 + 16);
      }
      // B fragments double-buffered: prefetch t+1 while computing on t,
      // each B fragment feeds two WMMAs (both tile-rows).
      FragU bf;
      {
        const _Float16* b0 = wsm + (qbase + lo) * LSTR + ks * 32 + hi * 8;
        bf.u[0] = *(const uint4*)(b0);  bf.u[1] = *(const uint4*)(b0 + 16);
      }
#pragma unroll
      for (int t = 0; t < 8; ++t) {
        FragU bfn;
        if (t < 7) {
          const _Float16* bn = wsm + (qbase + (t + 1) * 16 + lo) * LSTR + ks * 32 + hi * 8;
          bfn.u[0] = *(const uint4*)(bn);  bfn.u[1] = *(const uint4*)(bn + 16);
        }
        acc0[t] = __builtin_amdgcn_wmma_f32_16x16x32_f16(
            false, af0.v, false, bf.v, (short)0, acc0[t], false, false);
        acc1[t] = __builtin_amdgcn_wmma_f32_16x16x32_f16(
            false, af1.v, false, bf.v, (short)0, acc1[t], false, false);
        if (t < 7) bf = bfn;
      }
    }
  }

  __syncthreads();            // tiles dead; overlay result
  // ---- spill M (128x256 f32) to LDS: VGPR r of acc holds row r+8*hi, col lo ----
#pragma unroll
  for (int t = 0; t < 8; ++t)
#pragma unroll
    for (int r = 0; r < 8; ++r) {
      Ms[(tr0 * 16 + hi * 8 + r) * BSZ + qbase + t * 16 + lo]       = acc0[t][r];
      Ms[((tr0 + 1) * 16 + hi * 8 + r) * BSZ + qbase + t * 16 + lo] = acc1[t][r];
    }
  __syncthreads();

  // ---- row partials (axis A): 128 rows, one thread each ----
  if (tid < ROWS) {
    const float* mr = Ms + tid * BSZ;
    float m = -INFINITY;
    for (int c = 0; c < BSZ; ++c) m = fmaxf(m, mr[c]);
    float sm = 0.0f;
    for (int c = 0; c < BSZ; ++c) sm += __expf(mr[c] - m);
    float2 p = make_float2(m, sm);
    Apart[(size_t)s * BSZ + rowbase + tid] = p;
    rp[tid] = p;
  }
  __syncthreads();
  // ---- col partials (axis B): 256 cols, one thread each ----
  {
    float m = -INFINITY;
    for (int r = 0; r < ROWS; ++r) m = fmaxf(m, Ms[r * BSZ + tid]);
    float sm = 0.0f;
    for (int r = 0; r < ROWS; ++r) sm += __expf(Ms[r * BSZ + tid] - m);
    Bpart[((size_t)s * NPR + pr) * BSZ + tid] = make_float2(m, sm);
  }
  // ---- block partial (axis C): merge the 128 row pairs ----
  if (tid == 0) {
    float2 accc = make_float2(-INFINITY, 0.0f);
    for (int r = 0; r < ROWS; ++r) lse_merge(accc, rp[r].x, rp[r].y);
    Cpart[s * NPR + pr] = accc;
  }
}

// One block, 256 threads: thread a merges all partials for index a, adds exact
// f32 diagonal, block-reduces to the scalar loss.
__global__ __launch_bounds__(256)
void symile_finalize(const float* __restrict__ x, const float* __restrict__ y,
                     const float* __restrict__ z,
                     const float2* __restrict__ Apart, const float2* __restrict__ Bpart,
                     const float2* __restrict__ Cpart, float* __restrict__ out) {
  __shared__ float red[256];
  const int a = threadIdx.x;

  float2 aa = make_float2(-INFINITY, 0.0f);
  for (int s = 0; s < BSZ; ++s) { float2 p = Apart[(size_t)s * BSZ + a]; lse_merge(aa, p.x, p.y); }
  float lseA = aa.x + logf(aa.y);

  float2 bb = make_float2(-INFINITY, 0.0f);
  for (int i = 0; i < BSZ * NPR; ++i) { float2 p = Bpart[(size_t)i * BSZ + a]; lse_merge(bb, p.x, p.y); }
  float lseB = bb.x + logf(bb.y);

  float2 cc = make_float2(-INFINITY, 0.0f);
  for (int j = 0; j < NPR; ++j) { float2 p = Cpart[a * NPR + j]; lse_merge(cc, p.x, p.y); }
  float lseC = cc.x + logf(cc.y);

  float diag = 0.0f;
  const float* xr = x + (size_t)a * DDIM;
  const float* yr = y + (size_t)a * DDIM;
  const float* zr = z + (size_t)a * DDIM;
  for (int d = 0; d < DDIM; ++d) diag += xr[d] * yr[d] * zr[d];

  red[a] = lseA + lseB + lseC - 3.0f * diag;
  __syncthreads();
  for (int off = 128; off > 0; off >>= 1) {
    if (a < off) red[a] += red[a + off];
    __syncthreads();
  }
  if (a == 0) out[0] = red[0] * (1.0f / 768.0f);
}

extern "C" void kernel_launch(void* const* d_in, const int* in_sizes, int n_in,
                              void* d_out, int out_size, void* d_ws, size_t ws_size,
                              hipStream_t stream) {
  const float* x = (const float*)d_in[0];
  const float* y = (const float*)d_in[1];
  const float* z = (const float*)d_in[2];
  float* out = (float*)d_out;

  float2* Apart = (float2*)d_ws;                      // 256*256 pairs
  float2* Bpart = Apart + BSZ * BSZ;                  // 256*2*256 pairs
  float2* Cpart = Bpart + (size_t)BSZ * NPR * BSZ;    // 256*2 pairs

  dim3 grid(NPR, BSZ);
  symile_tile_kernel<<<grid, 256, SMEM_BYTES, stream>>>(x, y, z, Apart, Bpart, Cpart);
  symile_finalize<<<1, 256, 0, stream>>>(x, y, z, Apart, Bpart, Cpart, out);
}
